// Tagger_74251394614035
// MI455X (gfx1250) — matrix operand order repaired and checked by reading, not compiled, not run on hardware
//
#include <hip/hip_runtime.h>

// ---------------- problem constants ----------------
#define TT   128
#define BB   256
#define DD   300
#define DPAD 320          // D padded to multiple of 32 for WMMA K
#define H2C  256
#define G4   1024         // 4*H2
#define KTAG 64
#define START_TAG 62
#define STOP_TAG  63
#define MROWS (TT*BB)     // 32768

typedef __attribute__((ext_vector_type(16))) _Float16 v16h;
typedef __attribute__((ext_vector_type(8)))  _Float16 v8h;
typedef __attribute__((ext_vector_type(8)))  float    v8f;

// ---------------- WMMA fragment helpers (wave32, 16x16x32 f16 -> f32) -------
// A 16x32 (MxK): lane<16: M=lane, halves0-7=K0..7, halves8-15=K16..23
//                lane>=16: M=lane-16, halves0-7=K8..15, halves8-15=K24..31
// p must point at row base + k0 + (lane>>4)*8
__device__ __forceinline__ v16h frag_a_at(const _Float16* __restrict__ p) {
  v8h lo = *(const v8h*)p;
  v8h hi = *(const v8h*)(p + 16);
  return __builtin_shufflevector(lo, hi, 0,1,2,3,4,5,6,7,8,9,10,11,12,13,14,15);
}

// B 32x16 (KxN) from row-major weight W[N][K] (B = W^T):
// lane<16: N=lane, halves0-7=K0..7, halves8-15=K16..23 ; lane>=16: N=lane-16, K8..15, K24..31
// p must point at W-row base + k0 + (lane>>4)*8
__device__ __forceinline__ v16h frag_b_at(const _Float16* __restrict__ p) {
  v8h lo = *(const v8h*)p;
  v8h hi = *(const v8h*)(p + 16);
  return __builtin_shufflevector(lo, hi, 0,1,2,3,4,5,6,7,8,9,10,11,12,13,14,15);
}

// C/D f32 16x16: VGPR r: lane<16 -> (M=r, N=lane); lane>=16 -> (M=8+r, N=lane-16)
__device__ __forceinline__ v8f load_frag_c(const float* __restrict__ C,
                                           int ldc, int m0, int n0, int lane) {
  int n  = n0 + (lane & 15);
  int mb = m0 + ((lane >> 4) << 3);
  v8f r;
#pragma unroll
  for (int i = 0; i < 8; ++i) r[i] = C[(size_t)(mb + i) * ldc + n];
  return r;
}

__device__ __forceinline__ void store_frag(float* __restrict__ C, int ldc,
                                           int m0, int n0, v8f a, int lane) {
  int n  = n0 + (lane & 15);
  int mb = m0 + ((lane >> 4) << 3);
#pragma unroll
  for (int i = 0; i < 8; ++i) C[(size_t)(mb + i) * ldc + n] = a[i];
}

__device__ __forceinline__ v8f wmma_f16(v16h a, v16h b, v8f c) {
  return __builtin_amdgcn_wmma_f32_16x16x32_f16(false, a, false, b, (short)0, c,
                                                false, false);
}

// ---------------- prep: f32 -> f16 weights, bias folding, state init --------
__global__ void prep_kernel(const float* __restrict__ Wih_f, const float* __restrict__ Wih_b,
                            const float* __restrict__ Whh_f, const float* __restrict__ Whh_b,
                            const float* __restrict__ W_out,
                            const float* __restrict__ bih_f, const float* __restrict__ bhh_f,
                            const float* __restrict__ bih_b, const float* __restrict__ bhh_b,
                            const float* __restrict__ h0, const float* __restrict__ c0,
                            _Float16* __restrict__ Wih16_f, _Float16* __restrict__ Wih16_b,
                            _Float16* __restrict__ Whh16_f, _Float16* __restrict__ Whh16_b,
                            _Float16* __restrict__ Wout16,
                            float* __restrict__ biassum_f, float* __restrict__ biassum_b,
                            _Float16* __restrict__ hinit16, float* __restrict__ c_ws) {
  int gid = blockIdx.x * blockDim.x + threadIdx.x;
  int stride = gridDim.x * blockDim.x;
  for (int i = gid; i < G4 * DPAD; i += stride) {       // Wih padded 300->320
    int n = i / DPAD, k = i % DPAD;
    Wih16_f[i] = (_Float16)(k < DD ? Wih_f[n * DD + k] : 0.0f);
    Wih16_b[i] = (_Float16)(k < DD ? Wih_b[n * DD + k] : 0.0f);
  }
  for (int i = gid; i < G4 * H2C; i += stride) {
    Whh16_f[i] = (_Float16)Whh_f[i];
    Whh16_b[i] = (_Float16)Whh_b[i];
  }
  for (int i = gid; i < KTAG * 512; i += stride) Wout16[i] = (_Float16)W_out[i];
  for (int i = gid; i < G4; i += stride) {
    biassum_f[i] = bih_f[i] + bhh_f[i];
    biassum_b[i] = bih_b[i] + bhh_b[i];
  }
  for (int i = gid; i < 2 * BB * H2C; i += stride) {
    hinit16[i] = (_Float16)h0[i];
    c_ws[i]    = c0[i];
  }
}

// ---------------- embedding gather -> f16 X[MROWS][DPAD] --------------------
__global__ void gather_kernel(const int* __restrict__ sentences,
                              const float* __restrict__ word_emb,
                              _Float16* __restrict__ X) {
  size_t total = (size_t)MROWS * DPAD;
  size_t stride = (size_t)gridDim.x * blockDim.x;
  for (size_t idx = (size_t)blockIdx.x * blockDim.x + threadIdx.x; idx < total; idx += stride) {
    int m = (int)(idx / DPAD), k = (int)(idx % DPAD);
    int t = m / BB, b = m % BB;
    float v = 0.0f;
    if (k < DD) {
      int word = sentences[b * TT + t];
      v = word_emb[(size_t)word * DD + k];
    }
    X[idx] = (_Float16)v;
  }
}

// ---------------- generic WMMA GEMM: C[M][N] = A[M][K]*W[N][K]^T + bias -----
// grid: (N/64, M/256); block 128 threads = 4 waves; wave -> its own 64x64 tile
__global__ void __launch_bounds__(128, 1)
wmma_gemm_kernel(const _Float16* __restrict__ A, int lda,
                 const _Float16* __restrict__ W, int ldw,
                 const float* __restrict__ bias,
                 float* __restrict__ C, int ldc, int kblocks) {
  const int lane = threadIdx.x & 31;
  const int wave = threadIdx.x >> 5;
  const int n0 = blockIdx.x * 64;
  const int m0 = blockIdx.y * 256 + wave * 64;

  // per-lane base pointers, bumped by 32 along K each iteration
  const _Float16* ap[4];
  const _Float16* bp[4];
#pragma unroll
  for (int sm = 0; sm < 4; ++sm)
    ap[sm] = A + (size_t)(m0 + sm * 16 + (lane & 15)) * lda + ((lane >> 4) << 3);
#pragma unroll
  for (int sn = 0; sn < 4; ++sn)
    bp[sn] = W + (size_t)(n0 + sn * 16 + (lane & 15)) * ldw + ((lane >> 4) << 3);

  v8f acc[4][4];
#pragma unroll
  for (int sn = 0; sn < 4; ++sn) {
    float bv = bias[n0 + sn * 16 + (lane & 15)];
#pragma unroll
    for (int sm = 0; sm < 4; ++sm) {
#pragma unroll
      for (int i = 0; i < 8; ++i) acc[sm][sn][i] = bv;
    }
  }
  for (int kb = 0; kb < kblocks; ++kb) {
    v16h af[4];
#pragma unroll
    for (int sm = 0; sm < 4; ++sm) af[sm] = frag_a_at(ap[sm]);
#pragma unroll
    for (int sn = 0; sn < 4; ++sn) {
      v16h bf = frag_b_at(bp[sn]);
#pragma unroll
      for (int sm = 0; sm < 4; ++sm) acc[sm][sn] = wmma_f16(af[sm], bf, acc[sm][sn]);
    }
#pragma unroll
    for (int sm = 0; sm < 4; ++sm) ap[sm] += 32;
#pragma unroll
    for (int sn = 0; sn < 4; ++sn) bp[sn] += 32;
  }
#pragma unroll
  for (int sm = 0; sm < 4; ++sm)
#pragma unroll
    for (int sn = 0; sn < 4; ++sn)
      store_frag(C, ldc, m0 + sm * 16, n0 + sn * 16, acc[sm][sn], lane);
}

// ---------------- one LSTM time step (both directions) ----------------------
// grid: (B/64, H2/64, 2); block 128 = 4 waves, wave g computes gate g tile.
// gates = xW(t) [precomputed, biases folded] + h_prev * Whh^T; LDS exchange;
// fused cell update; h -> f16 concat buffer (backward stored time-reversed).
__global__ void __launch_bounds__(128, 1)
lstm_step_kernel(const _Float16* __restrict__ hprev_f,
                 const _Float16* __restrict__ hprev_b,
                 int lda_h,
                 const _Float16* __restrict__ Whh16_f,
                 const _Float16* __restrict__ Whh16_b,
                 const float* __restrict__ xw_f_t,
                 const float* __restrict__ xw_b_t,
                 float* __restrict__ c_f, float* __restrict__ c_b,
                 _Float16* __restrict__ hcat, int t) {
  __shared__ float lds[4][64][64];
  const int dir  = blockIdx.z;
  const int lane = threadIdx.x & 31;
  const int g    = threadIdx.x >> 5;   // gate = wave
  const int b0   = blockIdx.x * 64;
  const int j0   = blockIdx.y * 64;

  const _Float16* hprev = dir ? hprev_b : hprev_f;
  const _Float16* Whh   = dir ? Whh16_b : Whh16_f;
  const float*    xw    = dir ? xw_b_t : xw_f_t;
  float*          cbuf  = dir ? c_b : c_f;

  const _Float16* ap[4];
  const _Float16* bp[4];
#pragma unroll
  for (int sm = 0; sm < 4; ++sm)
    ap[sm] = hprev + (size_t)(b0 + sm * 16 + (lane & 15)) * lda_h + ((lane >> 4) << 3);
#pragma unroll
  for (int sn = 0; sn < 4; ++sn)
    bp[sn] = Whh + (size_t)(g * H2C + j0 + sn * 16 + (lane & 15)) * H2C + ((lane >> 4) << 3);

  v8f acc[4][4];
#pragma unroll
  for (int sm = 0; sm < 4; ++sm)
#pragma unroll
    for (int sn = 0; sn < 4; ++sn)
      acc[sm][sn] = load_frag_c(xw, G4, b0 + sm * 16, g * H2C + j0 + sn * 16, lane);

  for (int kb = 0; kb < H2C / 32; ++kb) {
    v16h af[4];
#pragma unroll
    for (int sm = 0; sm < 4; ++sm) af[sm] = frag_a_at(ap[sm]);
#pragma unroll
    for (int sn = 0; sn < 4; ++sn) {
      v16h bf = frag_b_at(bp[sn]);
#pragma unroll
      for (int sm = 0; sm < 4; ++sm) acc[sm][sn] = wmma_f16(af[sm], bf, acc[sm][sn]);
    }
#pragma unroll
    for (int sm = 0; sm < 4; ++sm) ap[sm] += 32;
#pragma unroll
    for (int sn = 0; sn < 4; ++sn) bp[sn] += 32;
  }
#pragma unroll
  for (int sm = 0; sm < 4; ++sm)
#pragma unroll
    for (int sn = 0; sn < 4; ++sn)
      store_frag(&lds[g][0][0], 64, sm * 16, sn * 16, acc[sm][sn], lane);
  __syncthreads();

  const int tt = dir ? (TT - 1 - t) : t;
  for (int e = threadIdx.x; e < 64 * 64; e += 128) {
    int bl = e >> 6, jl = e & 63;
    float iv = lds[0][bl][jl];
    float fv = lds[1][bl][jl];
    float gv = lds[2][bl][jl];
    float ov = lds[3][bl][jl];
    size_t ci = (size_t)(b0 + bl) * H2C + (j0 + jl);
    float c = cbuf[ci];
    float si = 1.0f / (1.0f + __expf(-iv));
    float sf = 1.0f / (1.0f + __expf(-fv));
    float so = 1.0f / (1.0f + __expf(-ov));
    float cn = sf * c + si * tanhf(gv);
    float hn = so * tanhf(cn);
    cbuf[ci] = cn;
    hcat[(size_t)tt * BB * 512 + (size_t)(b0 + bl) * 512 + dir * H2C + (j0 + jl)] =
        (_Float16)hn;
  }
}

// ---------------- CRF forward + gold score: one block per batch element -----
__global__ void crf_kernel(const float* __restrict__ feats,   // [T][B][K]
                           const float* __restrict__ trans,   // [K][K]
                           const int* __restrict__ tags,      // [B][T]
                           float* __restrict__ loss_b) {
  __shared__ float tr[KTAG][KTAG];
  __shared__ float fv[KTAG];
  __shared__ float red[KTAG];
  const int b = blockIdx.x;
  const int n = threadIdx.x;   // 64 threads
  for (int i = n; i < KTAG * KTAG; i += KTAG) tr[i >> 6][i & 63] = trans[i];
  fv[n] = (n == START_TAG) ? 0.0f : -10000.0f;
  __syncthreads();

  for (int t = 0; t < TT; ++t) {
    float ft = feats[((size_t)t * BB + b) * KTAG + n];
    float m = -1e30f;
    for (int p = 0; p < KTAG; ++p) m = fmaxf(m, fv[p] + tr[n][p]);
    float s = 0.0f;
    for (int p = 0; p < KTAG; ++p) s += __expf(fv[p] + tr[n][p] - m);
    float nf = ft + m + __logf(s);
    __syncthreads();
    fv[n] = nf;
    __syncthreads();
  }
  // alpha = logsumexp_p(fv[p] + trans[STOP][p])
  float v = fv[n] + tr[STOP_TAG][n];
  red[n] = v;
  __syncthreads();
  for (int off = 32; off > 0; off >>= 1) {
    if (n < off) red[n] = fmaxf(red[n], red[n + off]);
    __syncthreads();
  }
  float mx = red[0];
  __syncthreads();
  red[n] = __expf(v - mx);
  __syncthreads();
  for (int off = 32; off > 0; off >>= 1) {
    if (n < off) red[n] += red[n + off];
    __syncthreads();
  }
  float alpha = mx + __logf(red[0]);
  __syncthreads();

  // gold path score
  float gs = 0.0f;
  for (int t = n; t < TT; t += KTAG) {
    int tag  = tags[b * TT + t];
    int prev = (t == 0) ? START_TAG : tags[b * TT + t - 1];
    gs += tr[tag][prev] + feats[((size_t)t * BB + b) * KTAG + tag];
  }
  if (n == 0) gs += tr[STOP_TAG][tags[b * TT + TT - 1]];
  red[n] = gs;
  __syncthreads();
  for (int off = 32; off > 0; off >>= 1) {
    if (n < off) red[n] += red[n + off];
    __syncthreads();
  }
  if (n == 0) loss_b[b] = alpha - red[0];
}

__global__ void mean_kernel(const float* __restrict__ loss_b, float* __restrict__ out) {
  __shared__ float red[BB];
  int n = threadIdx.x;
  red[n] = loss_b[n];
  __syncthreads();
  for (int off = 128; off > 0; off >>= 1) {
    if (n < off) red[n] += red[n + off];
    __syncthreads();
  }
  if (n == 0) out[0] = red[0] / (float)BB;
}

// ---------------- host launcher ---------------------------------------------
extern "C" void kernel_launch(void* const* d_in, const int* in_sizes, int n_in,
                              void* d_out, int out_size, void* d_ws, size_t ws_size,
                              hipStream_t stream) {
  (void)in_sizes; (void)n_in; (void)out_size; (void)ws_size;

  const int*   sentences = (const int*)d_in[0];
  const int*   tags      = (const int*)d_in[1];
  const float* word_emb  = (const float*)d_in[2];
  const float* Wih_f     = (const float*)d_in[3];
  const float* Whh_f     = (const float*)d_in[4];
  const float* bih_f     = (const float*)d_in[5];
  const float* bhh_f     = (const float*)d_in[6];
  const float* Wih_b     = (const float*)d_in[7];
  const float* Whh_b     = (const float*)d_in[8];
  const float* bih_b     = (const float*)d_in[9];
  const float* bhh_b     = (const float*)d_in[10];
  const float* W_out     = (const float*)d_in[11];
  const float* b_out     = (const float*)d_in[12];
  const float* trans     = (const float*)d_in[13];
  const float* h0        = (const float*)d_in[14];
  const float* c0        = (const float*)d_in[15];

  // workspace carve
  char* w = (char*)d_ws;
  _Float16* X        = (_Float16*)w; w += (size_t)MROWS * DPAD * 2;       // 20.97 MB
  _Float16* Wih16_f  = (_Float16*)w; w += (size_t)G4 * DPAD * 2;
  _Float16* Wih16_b  = (_Float16*)w; w += (size_t)G4 * DPAD * 2;
  _Float16* Whh16_f  = (_Float16*)w; w += (size_t)G4 * H2C * 2;
  _Float16* Whh16_b  = (_Float16*)w; w += (size_t)G4 * H2C * 2;
  _Float16* Wout16   = (_Float16*)w; w += (size_t)KTAG * 512 * 2;
  float*    biassum_f= (float*)w;    w += (size_t)G4 * 4;
  float*    biassum_b= (float*)w;    w += (size_t)G4 * 4;
  _Float16* hinit16  = (_Float16*)w; w += (size_t)2 * BB * H2C * 2;
  float*    c_ws     = (float*)w;    w += (size_t)2 * BB * H2C * 4;       // c_f | c_b
  float*    xW_f     = (float*)w;    w += (size_t)MROWS * G4 * 4;         // 134 MB
  float*    xW_b     = (float*)w;    w += (size_t)MROWS * G4 * 4;         // 134 MB
  _Float16* hcat     = (_Float16*)w; w += (size_t)MROWS * 512 * 2;        // 33.5 MB
  float*    feats    = (float*)w;    w += (size_t)MROWS * KTAG * 4;       // 8.4 MB
  float*    loss_b   = (float*)w;    w += (size_t)BB * 4;
  float*    c_f = c_ws;
  float*    c_b = c_ws + (size_t)BB * H2C;

  prep_kernel<<<512, 256, 0, stream>>>(Wih_f, Wih_b, Whh_f, Whh_b, W_out,
                                       bih_f, bhh_f, bih_b, bhh_b, h0, c0,
                                       Wih16_f, Wih16_b, Whh16_f, Whh16_b, Wout16,
                                       biassum_f, biassum_b, hinit16, c_ws);
  gather_kernel<<<2048, 256, 0, stream>>>(sentences, word_emb, X);

  // input projections (biases folded): xW[t*B+b][n] = X @ Wih^T + (bih+bhh)
  dim3 gIn(G4 / 64, MROWS / 256);
  wmma_gemm_kernel<<<gIn, 128, 0, stream>>>(X, DPAD, Wih16_f, DPAD, biassum_f,
                                            xW_f, G4, DPAD / 32);
  wmma_gemm_kernel<<<gIn, 128, 0, stream>>>(X, DPAD, Wih16_b, DPAD, biassum_b,
                                            xW_b, G4, DPAD / 32);

  // recurrence: both directions per launch, 128 sequential launches
  for (int t = 0; t < TT; ++t) {
    const _Float16 *hpf, *hpb;
    int ldah;
    if (t == 0) {
      hpf = hinit16;
      hpb = hinit16 + (size_t)BB * H2C;
      ldah = H2C;
    } else {
      hpf = hcat + (size_t)(t - 1) * BB * 512;          // fwd h at time t-1
      hpb = hcat + (size_t)(TT - t) * BB * 512 + H2C;   // bwd h at time T-t
      ldah = 512;
    }
    lstm_step_kernel<<<dim3(4, 4, 2), 128, 0, stream>>>(
        hpf, hpb, ldah, Whh16_f, Whh16_b,
        xW_f + (size_t)t * BB * G4,
        xW_b + (size_t)(TT - 1 - t) * BB * G4,
        c_f, c_b, hcat, t);
  }

  // output projection: feats = hcat @ W_out^T + b_out
  dim3 gOut(KTAG / 64, MROWS / 256);
  wmma_gemm_kernel<<<gOut, 128, 0, stream>>>(hcat, 512, Wout16, 512, b_out,
                                             feats, KTAG, 512 / 32);

  crf_kernel<<<BB, KTAG, 0, stream>>>(feats, trans, tags, loss_b);
  mean_kernel<<<1, BB, 0, stream>>>(loss_b, (float*)d_out);
}
